// GraphMemoryVQ_1477468750581
// MI455X (gfx1250) — matrix operand order repaired and checked by reading, not compiled, not run on hardware
//
#include <hip/hip_runtime.h>
#include <hip/hip_bf16.h>

typedef __bf16 bf16;
typedef bf16  v16bf __attribute__((ext_vector_type(16)));
typedef bf16  v8bf  __attribute__((ext_vector_type(8)));
typedef float v8f   __attribute__((ext_vector_type(8)));

#define N_TOK  16384
#define KDIM   128      // 2*D
#define S_SYM  8192
#define ROWSTR 136      // bf16 elems per padded LDS row (272B: 16B-aligned, 4-bank stagger)
#define TILE_C 32       // codebook columns per iteration (2 wmma col-tiles)
#define BUFB   (TILE_C * ROWSTR * 2)   // bytes per LDS codebook buffer (8704)

// async global->LDS 16B copy (CDNA5, ASYNCcnt-tracked)
#define ASYNC_B128(lds, g) \
    asm volatile("global_load_async_to_lds_b128 %0, %1, off" :: "v"(lds), "v"(g) : "memory")
#define WAIT_ASYNC0() asm volatile("s_wait_asynccnt 0x0" ::: "memory")

// ---------------- kernel 0a: exact f32 codebook norms -> ws ----------------
__global__ __launch_bounds__(256) void vq_cnorm(const float* __restrict__ cb,
                                                float* __restrict__ cnorm) {
    int r = blockIdx.x * blockDim.x + threadIdx.x;
    if (r >= S_SYM) return;
    const float4* p = (const float4*)(cb + (size_t)r * KDIM);
    float s = 0.f;
#pragma unroll
    for (int i = 0; i < KDIM / 4; ++i) {
        float4 v = p[i];
        s += v.x * v.x + v.y * v.y + v.z * v.z + v.w * v.w;
    }
    cnorm[r] = s;
}

// ---------------- kernel 0b: codebook f32 -> bf16 (packed rows) ------------
__global__ __launch_bounds__(256) void vq_cvt_cb(const float* __restrict__ cb,
                                                 unsigned short* __restrict__ cb16) {
    size_t e = ((size_t)blockIdx.x * 256 + threadIdx.x) * 8;
    float4 a = *(const float4*)(cb + e);
    float4 b = *(const float4*)(cb + e + 4);
    union { bf16 h[8]; uint4 u; } pk;
    pk.h[0] = (bf16)a.x; pk.h[1] = (bf16)a.y; pk.h[2] = (bf16)a.z; pk.h[3] = (bf16)a.w;
    pk.h[4] = (bf16)b.x; pk.h[5] = (bf16)b.y; pk.h[6] = (bf16)b.z; pk.h[7] = (bf16)b.w;
    *(uint4*)(cb16 + e) = pk.u;
}

// ---------------- kernel 1: fused GEMM + bias + argmin --------------------
__global__ __launch_bounds__(256) void vq_argmin(
        const float* __restrict__ zr, const float* __restrict__ zi,
        const int*   __restrict__ prev,
        const unsigned short* __restrict__ cb16,   // bf16 codebook in ws
        const float* __restrict__ adj,
        const float* __restrict__ cnorm,
        float* __restrict__ out_idx) {
    __shared__ __align__(16) bf16 zt[8 * 16 * ROWSTR];        // per-wave z tiles
    __shared__ __align__(16) bf16 cbt[2][TILE_C * ROWSTR];    // double-buffered B tiles

    const int tid  = threadIdx.x;
    const int wave = tid >> 5;
    const int lane = tid & 31;
    const int lh   = lane >> 4;
    const int ln   = lane & 15;
    const int tok0 = blockIdx.x * 128 + wave * 16;

    bf16* zw = zt + wave * 16 * ROWSTR;

    // ---- per-thread async-copy slots: 2x 16B chunks per 32x128 bf16 tile ----
    // chunk c in [0,512): row = c>>4 (32 rows), col16 = c&15 (16B units)
    const int c0 = tid, c1 = tid + 256;
    const unsigned cbt_base = (unsigned)(uintptr_t)&cbt[0][0];
    unsigned lds0[2], lds1[2];
    lds0[0] = cbt_base + (unsigned)((c0 >> 4) * (ROWSTR * 2) + (c0 & 15) * 16);
    lds1[0] = cbt_base + (unsigned)((c1 >> 4) * (ROWSTR * 2) + (c1 & 15) * 16);
    lds0[1] = lds0[0] + BUFB;
    lds1[1] = lds1[0] + BUFB;
    unsigned long long g0 = (unsigned long long)(uintptr_t)cb16
                          + (unsigned long long)((c0 >> 4) * 256 + (c0 & 15) * 16);
    unsigned long long g1 = (unsigned long long)(uintptr_t)cb16
                          + (unsigned long long)((c1 >> 4) * 256 + (c1 & 15) * 16);

    // ---- kick off tile 0 into buffer 0 (overlaps the z-tile staging) ----
    ASYNC_B128(lds0[0], g0); g0 += TILE_C * 256;
    ASYNC_B128(lds1[0], g1); g1 += TILE_C * 256;

    // ---- stage this wave's 16x128 z_flat tile into LDS as bf16 ----
#pragma unroll
    for (int i = 0; i < 8; ++i) {
        int e = lane * 8 + i * 256;
        int r = e >> 7, c = e & 127;           // c multiple of 8
        const float* src = (c < 64) ? (zr + (size_t)(tok0 + r) * 64 + c)
                                    : (zi + (size_t)(tok0 + r) * 64 + (c - 64));
        float4 a = *(const float4*)src;
        float4 b = *(const float4*)(src + 4);
        union { bf16 h[8]; uint4 u; } pk;
        pk.h[0] = (bf16)a.x; pk.h[1] = (bf16)a.y; pk.h[2] = (bf16)a.z; pk.h[3] = (bf16)a.w;
        pk.h[4] = (bf16)b.x; pk.h[5] = (bf16)b.y; pk.h[6] = (bf16)b.z; pk.h[7] = (bf16)b.w;
        *(uint4*)&zw[r * ROWSTR + c] = pk.u;
    }

    // ---- hoist the 4 A operands into registers (DS ops in-order per wave) ----
    union { v16bf v; v8bf h[2]; } Au[4];
#pragma unroll
    for (int k0i = 0; k0i < 4; ++k0i) {
        int k0 = k0i * 32;
        Au[k0i].h[0] = *(const v8bf*)&zw[ln * ROWSTR + k0 + 8 * lh];
        Au[k0i].h[1] = *(const v8bf*)&zw[ln * ROWSTR + k0 + 16 + 8 * lh];
    }

    // ---- hoist adjacency row offsets: 8 per lane (loop-invariant) ----
    int aoffe[8];
#pragma unroll
    for (int v = 0; v < 8; ++v)
        aoffe[v] = prev[tok0 + v + 8 * lh] * S_SYM + ln;

    float bestv[8];
    int   besti[8];
#pragma unroll
    for (int v = 0; v < 8; ++v) { bestv[v] = 3.4e38f; besti[v] = 0; }

    WAIT_ASYNC0();
    __syncthreads();

    unsigned p = 0;
    for (int s0 = 0; s0 < S_SYM; s0 += TILE_C) {
        // prefetch next tile into the other buffer; overlaps compute below
        if (s0 + TILE_C < S_SYM) {
            ASYNC_B128(lds0[p ^ 1], g0); g0 += TILE_C * 256;
            ASYNC_B128(lds1[p ^ 1], g1); g1 += TILE_C * 256;
        }

        // ---- two independent 16x16 wmma chains sharing A ----
        const bf16* cbp = &cbt[p][0];
        v8f acc0 = {0.f, 0.f, 0.f, 0.f, 0.f, 0.f, 0.f, 0.f};
        v8f acc1 = {0.f, 0.f, 0.f, 0.f, 0.f, 0.f, 0.f, 0.f};
#pragma unroll
        for (int k0i = 0; k0i < 4; ++k0i) {
            int k0 = k0i * 32;
            union { v16bf v; v8bf h[2]; } B0, B1;
            B0.h[0] = *(const v8bf*)&cbp[ln * ROWSTR + k0 + 8 * lh];
            B0.h[1] = *(const v8bf*)&cbp[ln * ROWSTR + k0 + 16 + 8 * lh];
            B1.h[0] = *(const v8bf*)&cbp[(ln + 16) * ROWSTR + k0 + 8 * lh];
            B1.h[1] = *(const v8bf*)&cbp[(ln + 16) * ROWSTR + k0 + 16 + 8 * lh];
            acc0 = __builtin_amdgcn_wmma_f32_16x16x32_bf16(
                false, Au[k0i].v, false, B0.v, (short)0, acc0, false, false);
            acc1 = __builtin_amdgcn_wmma_f32_16x16x32_bf16(
                false, Au[k0i].v, false, B1.v, (short)0, acc1, false, false);
        }

        // ---- epilogue: ||c||^2 - 2*dot - 0.5*sigmoid(adj); running argmin ----
        float cn0 = cnorm[s0 + ln];
        float cn1 = cnorm[s0 + 16 + ln];
#pragma unroll
        for (int v = 0; v < 8; ++v) {
            float x0 = __builtin_nontemporal_load(&adj[(size_t)aoffe[v] + s0]);
            float x1 = __builtin_nontemporal_load(&adj[(size_t)aoffe[v] + s0 + 16]);
            float b0 = 0.5f * __builtin_amdgcn_rcpf(1.f + __expf(-x0));
            float b1 = 0.5f * __builtin_amdgcn_rcpf(1.f + __expf(-x1));
            float sc0 = cn0 - 2.f * acc0[v] - b0;
            float sc1 = cn1 - 2.f * acc1[v] - b1;
            if (sc0 < bestv[v]) { bestv[v] = sc0; besti[v] = s0 + ln; }
            if (sc1 < bestv[v]) { bestv[v] = sc1; besti[v] = s0 + 16 + ln; }
        }

        WAIT_ASYNC0();      // next tile resident before anyone flips buffers
        __syncthreads();
        p ^= 1;
    }

    // ---- cross-lane argmin reduction (reuse this wave's zt region) ----
    float* rv = (float*)zw;
    int*   ri = (int*)(rv + 256);
#pragma unroll
    for (int v = 0; v < 8; ++v) {
        int m = v + 8 * lh;
        rv[m * 16 + ln] = bestv[v];
        ri[m * 16 + ln] = besti[v];
    }
    __syncthreads();
    if (lane < 16) {
        int m = lane;
        float bv = rv[m * 16];
        int   bi = ri[m * 16];
        for (int c = 1; c < 16; ++c) {
            float v2 = rv[m * 16 + c];
            int   i2 = ri[m * 16 + c];
            if (v2 < bv) { bv = v2; bi = i2; }
        }
        out_idx[tok0 + m] = (float)bi;
    }
}

// ---------------- kernel 2: loss init ----------------
__global__ void vq_init_loss(float* loss) { *loss = 0.f; }

// ---------------- kernel 3: gather z_q, write outputs, loss ----------------
__global__ __launch_bounds__(256) void vq_gather_loss(
        const float* __restrict__ zr, const float* __restrict__ zi,
        const float* __restrict__ cb, const float* __restrict__ idxf,
        float* __restrict__ out_real, float* __restrict__ out_imag,
        float* __restrict__ loss) {
    __shared__ float red[256];
    int t = blockIdx.x * 2 + (threadIdx.x >> 7);
    int k = threadIdx.x & 127;
    int idx = (int)idxf[t];
    float c = cb[(size_t)idx * KDIM + k];
    float z = (k < 64) ? zr[(size_t)t * 64 + k] : zi[(size_t)t * 64 + (k - 64)];
    if (k < 64) out_real[(size_t)t * 64 + k] = c;
    else        out_imag[(size_t)t * 64 + (k - 64)] = c;
    float d = c - z;
    red[threadIdx.x] = d * d;
    __syncthreads();
    for (int s = 128; s > 0; s >>= 1) {
        if (threadIdx.x < s) red[threadIdx.x] += red[threadIdx.x + s];
        __syncthreads();
    }
    if (threadIdx.x == 0)
        atomicAdd(loss, red[0] * (1.25f / (float)(N_TOK * KDIM)));
}

extern "C" void kernel_launch(void* const* d_in, const int* in_sizes, int n_in,
                              void* d_out, int out_size, void* d_ws, size_t ws_size,
                              hipStream_t stream) {
    const float* zr   = (const float*)d_in[0];
    const float* zi   = (const float*)d_in[1];
    const int*   prev = (const int*)d_in[2];
    const float* cb   = (const float*)d_in[3];
    const float* adj  = (const float*)d_in[4];

    float* out      = (float*)d_out;
    float* out_real = out;                    // [N,64]
    float* out_imag = out + N_TOK * 64;       // [N,64]
    float* loss     = out + 2 * N_TOK * 64;   // [1]
    float* idxf     = loss + 1;               // [N] indices as float (exact)

    float*          cnorm = (float*)d_ws;                          // 32 KB
    unsigned short* cb16  = (unsigned short*)((char*)d_ws + 32768); // 2 MB bf16 codebook

    vq_cnorm<<<(S_SYM + 255) / 256, 256, 0, stream>>>(cb, cnorm);
    vq_cvt_cb<<<(S_SYM * KDIM / 8) / 256, 256, 0, stream>>>(cb, cb16);
    vq_init_loss<<<1, 1, 0, stream>>>(loss);
    vq_argmin<<<N_TOK / 128, 256, 0, stream>>>(zr, zi, prev, cb16, adj, cnorm, idxf);
    vq_gather_loss<<<N_TOK / 2, 256, 0, stream>>>(zr, zi, cb, idxf, out_real, out_imag, loss);
}
